// QCNN_Pure_12756052869697
// MI455X (gfx1250) — compile-verified, matrix operand order checked by reading, not run on hardware
//
#include <hip/hip_runtime.h>
#include <math.h>

// ---------------------------------------------------------------------------
// QCNN 8-qubit statevector simulator for gfx1250 (MI455X), wave32.
//
// Design: one batch element per wave. State = 256 complex amps:
//   amp index i[7:0]:  i[7:3] = lane id, i[2:0] = register slot r.
//   qubit w  <->  index bit (7 - w)   (row-major reshape in the reference).
// Gates on bits 0..2: in-register.  Bits 3..7: __shfl_xor lane exchange.
// Trig: hardware v_sin_f32/v_cos_f32 (__sinf/__cosf) — args bounded in
// [0, 4pi), avoids the branchy libm Payne-Hanek slow path entirely.
// Final <Z_7> reduction: v_wmma_f32_16x16x32_f16 with A = hi/lo f16 split of
// per-lane partials (error <= ~2^-22), B = ones, then 8 adds + 1 shfl_xor.
// ---------------------------------------------------------------------------

typedef __attribute__((ext_vector_type(16))) _Float16 v16h;
typedef __attribute__((ext_vector_type(8)))  float    v8f;

__device__ __forceinline__ float shflx(float v, int m) {
    return __shfl_xor(v, m, 32);
}

// Generic (optionally controlled) 1-qubit complex gate.
// QB / CB are *index bit positions* (0..7).  CB only used when HAS_C.
template <int QB, bool HAS_C, int CB>
__device__ __forceinline__ void apply1q(float* re, float* im, int lane,
                                        float u00r, float u00i, float u01r, float u01i,
                                        float u10r, float u10i, float u11r, float u11i)
{
    if (QB < 3) {                       // target bit lives in the register slot
        const int tm = 1 << QB;
#pragma unroll
        for (int r = 0; r < 8; ++r) {
            if (!(r & tm)) {
                const int r1 = r | tm;
                bool ctrl = true;
                if (HAS_C)
                    ctrl = (CB < 3) ? ((r & (1 << CB)) != 0)
                                    : ((lane & (1 << (CB - 3))) != 0);
                float a0r = re[r],  a0i = im[r];
                float a1r = re[r1], a1i = im[r1];
                float n0r = u00r*a0r - u00i*a0i + u01r*a1r - u01i*a1i;
                float n0i = u00r*a0i + u00i*a0r + u01r*a1i + u01i*a1r;
                float n1r = u10r*a0r - u10i*a0i + u11r*a1r - u11i*a1i;
                float n1i = u10r*a0i + u10i*a0r + u11r*a1i + u11i*a1r;
                re[r]  = ctrl ? n0r : a0r;  im[r]  = ctrl ? n0i : a0i;
                re[r1] = ctrl ? n1r : a1r;  im[r1] = ctrl ? n1i : a1i;
            }
        }
    } else {                            // target bit lives in the lane id
        const int lm = 1 << (QB - 3);
        const bool hiSide = (lane & lm) != 0;
#pragma unroll
        for (int r = 0; r < 8; ++r) {
            float ar = re[r], ai = im[r];
            float pr = shflx(ar, lm), pi = shflx(ai, lm);
            float a0r = hiSide ? pr : ar, a0i = hiSide ? pi : ai;
            float a1r = hiSide ? ar : pr, a1i = hiSide ? ai : pi;
            float lr = u00r*a0r - u00i*a0i + u01r*a1r - u01i*a1i;
            float li = u00r*a0i + u00i*a0r + u01r*a1i + u01i*a1r;
            float hr = u10r*a0r - u10i*a0i + u11r*a1r - u11i*a1i;
            float hi = u10r*a0i + u10i*a0r + u11r*a1i + u11i*a1r;
            float nr = hiSide ? hr : lr;
            float ni = hiSide ? hi : li;
            bool ctrl = true;
            if (HAS_C)
                ctrl = (CB < 3) ? ((r & (1 << CB)) != 0)
                                : ((lane & (1 << (CB - 3))) != 0);
            re[r] = ctrl ? nr : ar;  im[r] = ctrl ? ni : ai;
        }
    }
}

// CNOT: flip index bit BT where bit BC is set (pure permutation).
template <int BC, int BT>
__device__ __forceinline__ void cnotg(float* re, float* im, int lane)
{
    if (BC < 3 && BT < 3) {
#pragma unroll
        for (int r = 0; r < 8; ++r) {
            if ((r & (1 << BC)) && !(r & (1 << BT))) {
                const int r1 = r | (1 << BT);
                float tr = re[r], ti = im[r];
                re[r] = re[r1]; im[r] = im[r1];
                re[r1] = tr;    im[r1] = ti;
            }
        }
    } else if (BC < 3 && BT >= 3) {     // register-controlled lane swap: symmetric
        const int lm = 1 << (BT - 3);
#pragma unroll
        for (int r = 0; r < 8; ++r) {
            if (r & (1 << BC)) {
                re[r] = shflx(re[r], lm);
                im[r] = shflx(im[r], lm);
            }
        }
    } else if (BC >= 3 && BT < 3) {     // lane-controlled register swap
        const bool c = (lane & (1 << (BC - 3))) != 0;
#pragma unroll
        for (int r = 0; r < 8; ++r) {
            if (!(r & (1 << BT))) {
                const int r1 = r | (1 << BT);
                float a0r = re[r], a0i = im[r], a1r = re[r1], a1i = im[r1];
                re[r]  = c ? a1r : a0r;  im[r]  = c ? a1i : a0i;
                re[r1] = c ? a0r : a1r;  im[r1] = c ? a0i : a1i;
            }
        }
    } else {                            // lane-controlled lane swap: partners agree on BC
        const int lm = 1 << (BT - 3);
        const bool c = (lane & (1 << (BC - 3))) != 0;
#pragma unroll
        for (int r = 0; r < 8; ++r) {
            float pr = shflx(re[r], lm), pi = shflx(im[r], lm);
            re[r] = c ? pr : re[r];
            im[r] = c ? pi : im[r];
        }
    }
}

#define APPLY_U(BIT) apply1q<BIT, false, 0>(re, im, lane, \
    u00r, u00i, u01r, u01i, u10r, u10i, u11r, u11i)

// CRX(t) controlled on bit CB, target bit TB: RX = [[c,-is],[-is,c]]
#define APPLY_CRX(TB, CB, T) do {                                   \
    float gc = __cosf(0.5f * (T)), gs = __sinf(0.5f * (T));         \
    apply1q<TB, true, CB>(re, im, lane, gc, 0.f, 0.f, -gs,          \
                                        0.f, -gs, gc, 0.f);         \
} while (0)

// U3(t,p,l) on bit TB
#define APPLY_U3(TB, T, P, L) do {                                  \
    float ct = __cosf(0.5f * (T)), st = __sinf(0.5f * (T));         \
    float cl = __cosf(L), sl = __sinf(L);                           \
    float cp = __cosf(P), sp = __sinf(P);                           \
    float cpl = __cosf((P) + (L)), spl = __sinf((P) + (L));         \
    apply1q<TB, false, 0>(re, im, lane,                             \
        ct, 0.f,        -cl * st, -sl * st,                         \
        cp * st, sp * st,  cpl * ct, spl * ct);                     \
} while (0)

__global__ __launch_bounds__(256) void qcnn_kernel(
    const float* __restrict__ theta, const float* __restrict__ phi,
    const float* __restrict__ crx,   const float* __restrict__ u3p,
    float* __restrict__ out, int n)
{
    const int lane = threadIdx.x & 31;
    const int b = blockIdx.x * 8 + (threadIdx.x >> 5);   // one batch elem per wave
    if (b >= n) return;                                  // wave-uniform guard

    float re[8], im[8];
#pragma unroll
    for (int r = 0; r < 8; ++r) { re[r] = 0.f; im[r] = 0.f; }
    re[0] = (lane == 0) ? 1.f : 0.f;                     // |00000000>

    // Per-batch U = RZ(phi) @ RY(theta)
    const float th = theta[b], ph = phi[b];
    const float c  = __cosf(0.5f * th), s  = __sinf(0.5f * th);
    const float cf = __cosf(0.5f * ph), sf = __sinf(0.5f * ph);
    const float u00r =  cf * c, u00i = -sf * c;
    const float u01r = -cf * s, u01i =  sf * s;
    const float u10r =  cf * s, u10i =  sf * s;
    const float u11r =  cf * c, u11i =  sf * c;

    // ---- 4 cycles: U on every wire, then CNOT ring ----
#pragma unroll 1
    for (int cyc = 0; cyc < 4; ++cyc) {
        APPLY_U(7); APPLY_U(6); APPLY_U(5); APPLY_U(4);  // wires 0..3
        APPLY_U(3); APPLY_U(2); APPLY_U(1); APPLY_U(0);  // wires 4..7
        cnotg<7, 6>(re, im, lane);   // (0,1)
        cnotg<6, 5>(re, im, lane);   // (1,2)
        cnotg<5, 4>(re, im, lane);   // (2,3)
        cnotg<4, 3>(re, im, lane);   // (3,4)
        cnotg<3, 2>(re, im, lane);   // (4,5)
        cnotg<2, 1>(re, im, lane);   // (5,6)
        cnotg<1, 0>(re, im, lane);   // (6,7)
        cnotg<0, 7>(re, im, lane);   // (7,0)
    }

    // ---- QCNN layer 1: CRX (ctrl, tgt) with shared params ----
    APPLY_CRX(6, 7, crx[0]);   // (0,1)
    APPLY_CRX(4, 5, crx[1]);   // (2,3)
    APPLY_CRX(2, 3, crx[2]);   // (4,5)
    APPLY_CRX(0, 1, crx[3]);   // (6,7)
    APPLY_CRX(5, 6, crx[4]);   // (1,2)
    APPLY_CRX(3, 4, crx[5]);   // (3,4)
    APPLY_CRX(1, 2, crx[6]);   // (5,6)

    // U3 on wires 1,3,5,7  (bits 6,4,2,0)
    APPLY_U3(6, u3p[0], u3p[1], u3p[2]);
    APPLY_U3(4, u3p[3], u3p[4], u3p[5]);
    APPLY_U3(2, u3p[6], u3p[7], u3p[8]);
    APPLY_U3(0, u3p[9], u3p[10], u3p[11]);

    APPLY_CRX(4, 6, crx[7]);   // (1,3)
    APPLY_CRX(0, 2, crx[8]);   // (5,7)
    APPLY_CRX(2, 4, crx[9]);   // (3,5)

    APPLY_U3(4, u3p[12], u3p[13], u3p[14]);   // wire 3
    APPLY_U3(0, u3p[15], u3p[16], u3p[17]);   // wire 7

    APPLY_CRX(0, 4, crx[10]);  // (3,7)

    APPLY_U3(0, u3p[18], u3p[19], u3p[20]);   // wire 7

    // ---- <Z_7>: sign by index bit 0 (register slot parity) ----
    float p = 0.f;
#pragma unroll
    for (int r = 0; r < 8; ++r) {
        float pr = re[r] * re[r] + im[r] * im[r];
        p += (r & 1) ? -pr : pr;
    }

    // WMMA reduction stage: A[m,k] holds hi/lo f16 split of p for lane m
    // (lanes 0-15 -> K0/K1, lanes 16-31 -> K8/K9 per the documented 16-bit
    // A-matrix 16x32 layout), B = ones => D[m][n] = p[m] + p[m+16] for all n.
    _Float16 phi16 = (_Float16)p;
    float     plo  = p - (float)phi16;
    _Float16 plo16 = (_Float16)plo;

    v16h a = {};
    a[0] = phi16;
    a[1] = plo16;
    v16h bones;
#pragma unroll
    for (int k = 0; k < 16; ++k) bones[k] = (_Float16)1.0f;
    v8f cacc = {};
    cacc = __builtin_amdgcn_wmma_f32_16x16x32_f16(
        /*neg_a=*/false, a, /*neg_b=*/false, bones,
        /*c_mod=*/(short)0, cacc, /*reuse_a=*/false, /*reuse_b=*/false);

    // Per-lane sum of 8 accumulator rows gives sum of t[0..7] (lanes<16) or
    // t[8..15] (lanes>=16); one xor-16 exchange completes the 32-lane sum.
    float sum = cacc[0] + cacc[1] + cacc[2] + cacc[3] +
                cacc[4] + cacc[5] + cacc[6] + cacc[7];
    sum += shflx(sum, 16);

    if (lane == 0) out[b] = sum > 0.f ? sum : 0.f;       // relu
}

extern "C" void kernel_launch(void* const* d_in, const int* in_sizes, int n_in,
                              void* d_out, int out_size, void* d_ws, size_t ws_size,
                              hipStream_t stream) {
    const float* theta = (const float*)d_in[0];
    const float* phi   = (const float*)d_in[1];
    const float* crx   = (const float*)d_in[2];
    const float* u3p   = (const float*)d_in[3];
    float* out = (float*)d_out;
    const int n = in_sizes[0];               // B = 8192
    const int blocks = (n + 7) / 8;          // 8 waves (=8 batch elems) / block
    qcnn_kernel<<<blocks, 256, 0, stream>>>(theta, phi, crx, u3p, out, n);
}
